// GraphConvNet_2628519985616
// MI455X (gfx1250) — matrix operand order repaired and checked by reading, not compiled, not run on hardware
//
#include <hip/hip_runtime.h>
#include <math.h>

#define LATENT   128
#define F_IN     16
#define NGRAPH   128
#define OUTG     10
#define NSTEPS   3
#define LN_EPS_F 1e-6f

typedef __attribute__((ext_vector_type(16))) __bf16         bf16x16;
typedef __attribute__((ext_vector_type(16))) unsigned short u16x16;
typedef __attribute__((ext_vector_type(8)))  unsigned short u16x8;
typedef __attribute__((ext_vector_type(8)))  float          f32x8;

__device__ __forceinline__ unsigned short f2bf(float f) {
    // round-to-nearest-even fp32 -> bf16 bit pattern
    unsigned int u = __float_as_uint(f);
    u += 0x7FFFu + ((u >> 16) & 1u);
    return (unsigned short)(u >> 16);
}

__device__ __forceinline__ float fast_tanhf(float x) {
#if __has_builtin(__builtin_amdgcn_tanhf)
    return __builtin_amdgcn_tanhf(x);   // V_TANH_F32
#else
    return tanhf(x);
#endif
}

// ---------------------------------------------------------------- utilities
// Convert MLP weights fp32 [mat][k][n] -> bf16 TRANSPOSED [mat][n][k]
// (so WMMA B-operand per-lane slices are contiguous in LDS)
__global__ void convert_bf16_T_kernel(const float* __restrict__ src,
                                      unsigned short* __restrict__ dst, int total) {
    int i = blockIdx.x * blockDim.x + threadIdx.x;
    if (i >= total) return;
    int mat = i / (LATENT * LATENT);
    int rem = i - mat * (LATENT * LATENT);
    int n = rem >> 7;   // dst row  = col of W
    int k = rem & 127;  // dst col  = row of W
    dst[i] = f2bf(src[(size_t)mat * LATENT * LATENT + k * LATENT + n]);
}

__global__ void zero_f32_kernel(float* __restrict__ p, long n) {
    long i = (long)blockIdx.x * blockDim.x + threadIdx.x;
    long stride = (long)gridDim.x * blockDim.x;
    for (; i < n; i += stride) p[i] = 0.0f;
}

__global__ void degree_kernel(const int* __restrict__ s, const int* __restrict__ r,
                              int* __restrict__ degS, int* __restrict__ degR, int nE) {
    int e = blockIdx.x * blockDim.x + threadIdx.x;
    if (e < nE) {
        atomicAdd(&degS[s[e]], 1);
        atomicAdd(&degR[r[e]], 1);
    }
}

__global__ void invdeg_kernel(const int* __restrict__ degS, const int* __restrict__ degR,
                              float* __restrict__ invS, float* __restrict__ invR, int n) {
    int i = blockIdx.x * blockDim.x + threadIdx.x;
    if (i < n) {
        // +1 for the self edge; degree >= 1 always, so max(.,1) is implicit
        invS[i] = rsqrtf((float)(degS[i] + 1));
        invR[i] = rsqrtf((float)(degR[i] + 1));
    }
}

// ---------------------------------------------------------------- embedder
__global__ void embed_kernel(const float* __restrict__ x, const float* __restrict__ We,
                             const float* __restrict__ be, float* __restrict__ h) {
    int node = blockIdx.x;
    int j = threadIdx.x;
    const float* xr = x + (size_t)node * F_IN;
    float acc = be[j];
#pragma unroll
    for (int k = 0; k < F_IN; ++k)
        acc = fmaf(xr[k], We[k * LATENT + j], acc);
    h[(size_t)node * LATENT + j] = acc;
}

// ---------------------------------------------------------------- fused 2-layer MLP via WMMA
// u_out[n,:] = tanh(tanh(h[n,:] @ W1 + b1) @ W2 + b2) * invS[n]
// Block = 256 threads (8 waves), 16 nodes per block; wave w owns cols [16w,16w+16).
// W1t/W2t are bf16 bit patterns, TRANSPOSED: Wt[n][k] = W[k][n].
__global__ void __launch_bounds__(256)
mlp_wmma_kernel(const float* __restrict__ h,
                const unsigned short* __restrict__ W1t, const float* __restrict__ b1,
                const unsigned short* __restrict__ W2t, const float* __restrict__ b2,
                const float* __restrict__ invS, float* __restrict__ u_out, int nNodes) {
    __shared__ __align__(32) unsigned short sA[16 * LATENT];      // activations
    __shared__ __align__(32) unsigned short sB[LATENT * LATENT];  // W^T tile
    __shared__ __align__(32) unsigned short sO[16 * LATENT];      // layer-1 output

    const int t = threadIdx.x;
    const int base = blockIdx.x * 16;

    // stage A tile: thread t converts 8 consecutive floats (2 x float4)
    {
        int row = t >> 4;               // 16 threads per row
        int colx = (t & 15) * 8;
        int node = base + row;
        if (node >= nNodes) node = nNodes - 1;
        const float4* hp = (const float4*)(h + (size_t)node * LATENT + colx);
        float4 f0 = hp[0];
        float4 f1 = hp[1];
        u16x8 p;
        p[0] = f2bf(f0.x); p[1] = f2bf(f0.y); p[2] = f2bf(f0.z); p[3] = f2bf(f0.w);
        p[4] = f2bf(f1.x); p[5] = f2bf(f1.y); p[6] = f2bf(f1.z); p[7] = f2bf(f1.w);
        *(u16x8*)(sA + t * 8) = p;
    }
    // stage W1^T: 2048 uint4 total, 8 per thread (b128 loads/stores)
    {
        const uint4* src = (const uint4*)W1t;
        uint4* dst = (uint4*)sB;
#pragma unroll
        for (int i = 0; i < 8; ++i)
            dst[i * 256 + t] = src[i * 256 + t];
    }
    // warm L2/L0 with W2^T while layer 1 computes (global_prefetch_b8)
    __builtin_prefetch(W2t + (size_t)t * 64, 0, 3);
    __syncthreads();

    const int lane = t & 31;
    const int wave = t >> 5;
    const int col  = wave * 16 + (lane & 15);
    const int rowA = lane & 15;
    const int kAoff = (lane >= 16) ? 16 : 0;   // A: K = {0..15} vs {16..31}
    const int kBoff = (lane >= 16) ? 16 : 0;   // B: contiguous 16-K slice per half-wave
    const int mOff  = (lane >= 16) ? 8 : 0;    // C/D: VGPR i -> row i (+8 for hi lanes)

    // ---------------- layer 1 ----------------
    f32x8 c;
    {
        float bv = b1[col];
#pragma unroll
        for (int i = 0; i < 8; ++i) c[i] = bv;
    }
#pragma unroll
    for (int kt = 0; kt < LATENT / 32; ++kt) {
        const int ka = kt * 32 + kAoff;
        const int kb = kt * 32 + kBoff;
        u16x8 alo = *(const u16x8*)(sA + rowA * LATENT + ka);
        u16x8 ahi = *(const u16x8*)(sA + rowA * LATENT + ka + 8);
        u16x16 au = __builtin_shufflevector(alo, ahi, 0, 1, 2, 3, 4, 5, 6, 7,
                                            8, 9, 10, 11, 12, 13, 14, 15);
        u16x16 bu = *(const u16x16*)(sB + col * LATENT + kb);  // 32B aligned, contiguous
        bf16x16 a = __builtin_bit_cast(bf16x16, au);
        bf16x16 b = __builtin_bit_cast(bf16x16, bu);
        c = __builtin_amdgcn_wmma_f32_16x16x32_bf16(false, a, false, b, (short)0, c,
                                                    false, false);
    }
#pragma unroll
    for (int i = 0; i < 8; ++i)
        sO[(i + mOff) * LATENT + col] = f2bf(fast_tanhf(c[i]));
    __syncthreads();

    // stage W2^T over W1^T
    {
        const uint4* src = (const uint4*)W2t;
        uint4* dst = (uint4*)sB;
#pragma unroll
        for (int i = 0; i < 8; ++i)
            dst[i * 256 + t] = src[i * 256 + t];
    }
    __syncthreads();

    // ---------------- layer 2 ----------------
    {
        float bv = b2[col];
#pragma unroll
        for (int i = 0; i < 8; ++i) c[i] = bv;
    }
#pragma unroll
    for (int kt = 0; kt < LATENT / 32; ++kt) {
        const int ka = kt * 32 + kAoff;
        const int kb = kt * 32 + kBoff;
        u16x8 alo = *(const u16x8*)(sO + rowA * LATENT + ka);
        u16x8 ahi = *(const u16x8*)(sO + rowA * LATENT + ka + 8);
        u16x16 au = __builtin_shufflevector(alo, ahi, 0, 1, 2, 3, 4, 5, 6, 7,
                                            8, 9, 10, 11, 12, 13, 14, 15);
        u16x16 bu = *(const u16x16*)(sB + col * LATENT + kb);
        bf16x16 a = __builtin_bit_cast(bf16x16, au);
        bf16x16 b = __builtin_bit_cast(bf16x16, bu);
        c = __builtin_amdgcn_wmma_f32_16x16x32_bf16(false, a, false, b, (short)0, c,
                                                    false, false);
    }
#pragma unroll
    for (int i = 0; i < 8; ++i) {
        int node = base + i + mOff;
        if (node < nNodes)
            u_out[(size_t)node * LATENT + col] = fast_tanhf(c[i]) * invS[node];
    }
}

// ---------------------------------------------------------------- edge scatter
// agg[r[e], :] += u[s[e], :]   (one wave per edge, float4 per lane)
__global__ void scatter_kernel(const int* __restrict__ s, const int* __restrict__ r,
                               const float* __restrict__ u, float* __restrict__ agg,
                               int nE) {
    long tid = (long)blockIdx.x * blockDim.x + threadIdx.x;
    int e = (int)(tid >> 5);
    int lane = (int)(tid & 31);
    if (e >= nE) return;
    int sn = s[e];
    int rn = r[e];
    const float4 v = *(const float4*)(u + (size_t)sn * LATENT + lane * 4);
    float* dst = agg + (size_t)rn * LATENT + lane * 4;
    atomicAdd(dst + 0, v.x);
    atomicAdd(dst + 1, v.y);
    atomicAdd(dst + 2, v.z);
    atomicAdd(dst + 3, v.w);
}

// ---------------------------------------------------------------- skip + layernorm
// h[n,:] = LN((agg[n,:] + u[n,:]) * invR[n] + h[n,:])  (self-edge folded in as +u[n,:])
__global__ void combine_ln_kernel(const float* __restrict__ agg, const float* __restrict__ u,
                                  const float* __restrict__ invR, float* __restrict__ h,
                                  const float* __restrict__ scale,
                                  const float* __restrict__ bias) {
    __shared__ float red[LATENT];
    int n = blockIdx.x;
    int j = threadIdx.x;
    size_t idx = (size_t)n * LATENT + j;
    float val = (agg[idx] + u[idx]) * invR[n] + h[idx];

    red[j] = val;
    __syncthreads();
    for (int off = LATENT / 2; off > 0; off >>= 1) {
        if (j < off) red[j] += red[j + off];
        __syncthreads();
    }
    float mu = red[0] * (1.0f / LATENT);
    __syncthreads();
    float d = val - mu;
    red[j] = d * d;
    __syncthreads();
    for (int off = LATENT / 2; off > 0; off >>= 1) {
        if (j < off) red[j] += red[j + off];
        __syncthreads();
    }
    float var = red[0] * (1.0f / LATENT);
    h[idx] = d * rsqrtf(var + LN_EPS_F) * scale[j] + bias[j];
}

// ---------------------------------------------------------------- pooling + decode
__global__ void pool_kernel(const float* __restrict__ h, const int* __restrict__ gid,
                            float* __restrict__ pooled, float* __restrict__ cnt) {
    int n = blockIdx.x;
    int j = threadIdx.x;
    int g = gid[n];
    atomicAdd(&pooled[(size_t)g * LATENT + j], h[(size_t)n * LATENT + j]);
    if (j == 0) atomicAdd(&cnt[g], 1.0f);
}

__global__ void decode_kernel(const float* __restrict__ pooled, const float* __restrict__ cnt,
                              const float* __restrict__ Wd, const float* __restrict__ bd,
                              float* __restrict__ out) {
    int g = blockIdx.x;
    int o = threadIdx.x;
    if (o >= OUTG) return;
    float inv = 1.0f / fmaxf(cnt[g], 1.0f);
    float acc = bd[o];
#pragma unroll 8
    for (int j = 0; j < LATENT; ++j)
        acc = fmaf(pooled[(size_t)g * LATENT + j] * inv, Wd[j * OUTG + o], acc);
    out[g * OUTG + o] = acc;
}

// ---------------------------------------------------------------- host launcher
extern "C" void kernel_launch(void* const* d_in, const int* in_sizes, int n_in,
                              void* d_out, int out_size, void* d_ws, size_t ws_size,
                              hipStream_t stream) {
    const float* x        = (const float*)d_in[0];
    const int*   senders  = (const int*)d_in[1];
    const int*   receivers= (const int*)d_in[2];
    const int*   gid      = (const int*)d_in[3];
    const float* W_embed  = (const float*)d_in[4];
    const float* b_embed  = (const float*)d_in[5];
    const float* W_mlp    = (const float*)d_in[6];
    const float* b_mlp    = (const float*)d_in[7];
    const float* ln_scale = (const float*)d_in[8];
    const float* ln_bias  = (const float*)d_in[9];
    const float* W_dec    = (const float*)d_in[10];
    const float* b_dec    = (const float*)d_in[11];
    float* out = (float*)d_out;

    const int N = in_sizes[3];   // graph_ids length == N_NODES
    const int E = in_sizes[1];   // senders length == N_EDGES

    // workspace carve-out (256B aligned slices)
    char* ws = (char*)d_ws;
    size_t off = 0;
    auto carve = [&](size_t bytes) -> char* {
        char* p = ws + off;
        off += (bytes + 255) & ~(size_t)255;
        return p;
    };
    float* h   = (float*)carve((size_t)N * LATENT * sizeof(float));
    float* u   = (float*)carve((size_t)N * LATENT * sizeof(float));
    float* agg = (float*)carve((size_t)N * LATENT * sizeof(float));
    unsigned short* Wb = (unsigned short*)carve((size_t)NSTEPS * 2 * LATENT * LATENT * sizeof(unsigned short));
    int* deg   = (int*)carve((size_t)2 * N * sizeof(int));   // degS | degR contiguous
    int* degS  = deg;
    int* degR  = deg + N;
    float* invS = (float*)carve((size_t)N * sizeof(float));
    float* invR = (float*)carve((size_t)N * sizeof(float));
    float* pooled = (float*)carve((size_t)(NGRAPH * LATENT + NGRAPH) * sizeof(float));
    float* cnt    = pooled + NGRAPH * LATENT;

    // 1) weights -> bf16, transposed per matrix
    const int nw = NSTEPS * 2 * LATENT * LATENT;
    convert_bf16_T_kernel<<<(nw + 255) / 256, 256, 0, stream>>>(W_mlp, Wb, nw);

    // 2) degrees + inv-sqrt (with self edge)
    zero_f32_kernel<<<256, 256, 0, stream>>>((float*)deg, (long)2 * N);
    degree_kernel<<<(E + 255) / 256, 256, 0, stream>>>(senders, receivers, degS, degR, E);
    invdeg_kernel<<<(N + 255) / 256, 256, 0, stream>>>(degS, degR, invS, invR, N);

    // 3) embed
    embed_kernel<<<N, LATENT, 0, stream>>>(x, W_embed, b_embed, h);

    // 4) message-passing steps
    for (int t = 0; t < NSTEPS; ++t) {
        const unsigned short* W1t = Wb + (size_t)(t * 2 + 0) * LATENT * LATENT;
        const unsigned short* W2t = Wb + (size_t)(t * 2 + 1) * LATENT * LATENT;
        const float* b1 = b_mlp + (size_t)(t * 2 + 0) * LATENT;
        const float* b2 = b_mlp + (size_t)(t * 2 + 1) * LATENT;

        mlp_wmma_kernel<<<(N + 15) / 16, 256, 0, stream>>>(h, W1t, b1, W2t, b2, invS, u, N);
        zero_f32_kernel<<<1024, 256, 0, stream>>>(agg, (long)N * LATENT);
        scatter_kernel<<<(int)(((long)E * 32 + 255) / 256), 256, 0, stream>>>(senders, receivers, u, agg, E);
        combine_ln_kernel<<<N, LATENT, 0, stream>>>(agg, u, invR, h,
                                                    ln_scale + t * LATENT,
                                                    ln_bias + t * LATENT);
    }

    // 5) segment-mean pool + decode
    zero_f32_kernel<<<64, 256, 0, stream>>>(pooled, (long)(NGRAPH * LATENT + NGRAPH));
    pool_kernel<<<N, LATENT, 0, stream>>>(h, gid, pooled, cnt);
    decode_kernel<<<NGRAPH, 32, 0, stream>>>(pooled, cnt, W_dec, b_dec, out);
}
